// MultiHeadAttention_18124761989656
// MI455X (gfx1250) — matrix operand order repaired and checked by reading, not compiled, not run on hardware
//
#include <hip/hip_runtime.h>

#define SEQ     2048
#define DMODEL  1024
#define NHEADS  16
#define DHEAD   64
#define NBATCH  2
#define NKROWS  (NBATCH * SEQ)   // 4096

typedef __attribute__((ext_vector_type(16))) __bf16 v16bf;
typedef __attribute__((ext_vector_type(8)))  float  v8f;

union Frag16 { v16bf v; uint4 q[2]; };

__device__ __forceinline__ unsigned short f2bf(float f) {
  __bf16 h = (__bf16)f;                       // native v_cvt on gfx1250
  return __builtin_bit_cast(unsigned short, h);
}
__device__ __forceinline__ unsigned pack2(float a, float b) {
  return (unsigned)f2bf(a) | ((unsigned)f2bf(b) << 16);
}

#define WMMA_BF16(A, B, C) \
  __builtin_amdgcn_wmma_f32_16x16x32_bf16(false, (A), false, (B), (short)0, (C), false, false)

// ---------------------------------------------------------------------------
// Kernel 0: elementwise f32 -> bf16 (weights pre-conversion, bandwidth bound)
// ---------------------------------------------------------------------------
__global__ __launch_bounds__(256)
void cvt_bf16_kernel(const float* __restrict__ in, unsigned short* __restrict__ out, int n4) {
  int i = blockIdx.x * 256 + threadIdx.x;
  if (i < n4) {
    float4 f = ((const float4*)in)[i];
    uint2 p; p.x = pack2(f.x, f.y); p.y = pack2(f.z, f.w);
    ((uint2*)out)[i] = p;
  }
}

// ---------------------------------------------------------------------------
// Kernel 1: per-head linear projection.  out[h][row][d] = (X[row]·W[h][d] + b)*scale
// W pre-converted bf16.  storeT==1 stores transposed [h][n][d][k] (for V).
// Block = 128 threads (4 waves); wave w computes d-tile w*16.
// ---------------------------------------------------------------------------
__global__ __launch_bounds__(128)
void qkv_proj_kernel(const float* __restrict__ X, const unsigned short* __restrict__ W,
                     const float* __restrict__ bias, unsigned short* __restrict__ out,
                     float outScale, int storeT) {
  __shared__ unsigned short ldsX[16 * 1032];   // 16 rows x 1024 bf16, +8 pad/row
  const int tid  = threadIdx.x;
  const int row0 = blockIdx.x * 16;
  const int h    = blockIdx.y;

  const float* xt = X + (size_t)row0 * DMODEL;
  for (int i = 0; i < 32; ++i) {
    int c   = tid + i * 128;
    int row = c >> 8;
    int col = (c & 255) << 2;
    float4 f = *(const float4*)(xt + row * DMODEL + col);
    uint2 p; p.x = pack2(f.x, f.y); p.y = pack2(f.z, f.w);
    *(uint2*)(&ldsX[row * 1032 + col]) = p;
  }
  __syncthreads();

  const int wave = tid >> 5;
  const int lane = tid & 31;
  const int ln = lane & 15, g = lane >> 4;
  const int d0 = wave * 16;
  const int j  = d0 + ln;

  const unsigned short* wrow = W + ((size_t)h * DHEAD + j) * DMODEL;
  const unsigned short* arow = &ldsX[ln * 1032];

  v8f acc = {0.f,0.f,0.f,0.f,0.f,0.f,0.f,0.f};
  #pragma unroll 4
  for (int k0 = 0; k0 < DMODEL; k0 += 32) {
    Frag16 a, b;
    a.q[0] = *(const uint4*)(arow + k0 + 8 * g);
    a.q[1] = *(const uint4*)(arow + k0 + 16 + 8 * g);
    b.q[0] = *(const uint4*)(wrow + k0 + 8 * g);
    b.q[1] = *(const uint4*)(wrow + k0 + 16 + 8 * g);
    acc = WMMA_BF16(a.v, b.v, acc);
  }

  const float bv = bias[h * DHEAD + j];
  if (!storeT) {                                         // Q/K: [H][NK][64]
    unsigned short* orow = out + ((size_t)h * NKROWS + row0) * DHEAD + j;
    #pragma unroll
    for (int r = 0; r < 8; ++r)
      orow[(size_t)(r + 8 * g) * DHEAD] = f2bf((acc[r] + bv) * outScale);
  } else {                                               // V: [H][N][64][SEQ]
    int n  = row0 >> 11;
    int kb = row0 & (SEQ - 1);
    uint4 pk;
    pk.x = pack2((acc[0]+bv)*outScale, (acc[1]+bv)*outScale);
    pk.y = pack2((acc[2]+bv)*outScale, (acc[3]+bv)*outScale);
    pk.z = pack2((acc[4]+bv)*outScale, (acc[5]+bv)*outScale);
    pk.w = pack2((acc[6]+bv)*outScale, (acc[7]+bv)*outScale);
    *(uint4*)(out + (((size_t)h * NBATCH + n) * DHEAD + j) * SEQ + kb + 8 * g) = pk;
  }
}

// ---------------------------------------------------------------------------
// Kernel 2: flash attention, 64-key blocks.  One wave owns 16 query rows of
// one (h,n); online softmax once per 64 keys (halved shuffle/rescale rate).
// Q pre-scaled by 1/sqrt(DHEAD).  8 waves/block.
// ---------------------------------------------------------------------------
#define PSTRIDE 68   // 16x64 f32 P staging row stride (+4 pad)

__global__ __launch_bounds__(256)
void attn_kernel(const unsigned short* __restrict__ Qb,
                 const unsigned short* __restrict__ Kb,
                 const unsigned short* __restrict__ Vt,
                 unsigned short* __restrict__ Y) {
  __shared__ float ldsP[8 * 16 * PSTRIDE];   // 34.8 KB
  const int tid  = threadIdx.x;
  const int wave = tid >> 5, lane = tid & 31;
  const int ln = lane & 15, g = lane >> 4;

  const int flat = blockIdx.x * 8 + wave;    // 0..4095 = H*N*(SEQ/16)
  const int h  = flat >> 8;
  const int n  = (flat >> 7) & 1;
  const int q0 = (flat & 127) << 4;

  float* myP = &ldsP[wave * (16 * PSTRIDE)];

  const unsigned short* qrow = Qb + ((size_t)h * NKROWS + n * SEQ + q0 + ln) * DHEAD;
  Frag16 qa0, qa1;
  qa0.q[0] = *(const uint4*)(qrow + 8 * g);
  qa0.q[1] = *(const uint4*)(qrow + 16 + 8 * g);
  qa1.q[0] = *(const uint4*)(qrow + 32 + 8 * g);
  qa1.q[1] = *(const uint4*)(qrow + 48 + 8 * g);

  const unsigned short* kbase = Kb + ((size_t)h * NKROWS + n * SEQ) * DHEAD;
  const unsigned short* vbase = Vt + ((size_t)h * NBATCH + n) * DHEAD * (size_t)SEQ;

  float mrow[8], lrow[8];
  v8f o[4] = {};
  #pragma unroll
  for (int r = 0; r < 8; ++r) { mrow[r] = -1e30f; lrow[r] = 0.f; }

  const float L2E = 1.4426950408889634f;

  for (int l0 = 0; l0 < SEQ; l0 += 64) {
    // ---- S = Q·K^T for 4 column tiles (64 keys) ----
    v8f s[4];
    #pragma unroll
    for (int i = 0; i < 4; ++i) {
      const unsigned short* kr = kbase + (size_t)(l0 + 16 * i + ln) * DHEAD + 8 * g;
      Frag16 b;
      v8f si = {0.f,0.f,0.f,0.f,0.f,0.f,0.f,0.f};
      b.q[0] = *(const uint4*)(kr);      b.q[1] = *(const uint4*)(kr + 16);
      si = WMMA_BF16(qa0.v, b.v, si);
      b.q[0] = *(const uint4*)(kr + 32); b.q[1] = *(const uint4*)(kr + 48);
      si = WMMA_BF16(qa1.v, b.v, si);
      s[i] = si;
    }
    // ---- online softmax over 64 columns ----
    #pragma unroll
    for (int r = 0; r < 8; ++r) {
      float v0 = s[0][r], v1 = s[1][r], v2 = s[2][r], v3 = s[3][r];
      float mx = fmaxf(fmaxf(v0, v1), fmaxf(v2, v3));
      mx = fmaxf(mx, __shfl_xor(mx, 1, 32));
      mx = fmaxf(mx, __shfl_xor(mx, 2, 32));
      mx = fmaxf(mx, __shfl_xor(mx, 4, 32));
      mx = fmaxf(mx, __shfl_xor(mx, 8, 32));
      float mnew = fmaxf(mrow[r], mx);
      float fac = exp2f((mrow[r] - mnew) * L2E);
      float p0 = exp2f((v0 - mnew) * L2E);
      float p1 = exp2f((v1 - mnew) * L2E);
      float p2 = exp2f((v2 - mnew) * L2E);
      float p3 = exp2f((v3 - mnew) * L2E);
      float rs = (p0 + p1) + (p2 + p3);
      rs += __shfl_xor(rs, 1, 32);
      rs += __shfl_xor(rs, 2, 32);
      rs += __shfl_xor(rs, 4, 32);
      rs += __shfl_xor(rs, 8, 32);
      lrow[r] = lrow[r] * fac + rs;
      mrow[r] = mnew;
      o[0][r] *= fac; o[1][r] *= fac; o[2][r] *= fac; o[3][r] *= fac;
      float* pr = myP + (r + 8 * g) * PSTRIDE + ln;   // C-layout -> A-layout
      pr[0] = p0; pr[16] = p1; pr[32] = p2; pr[48] = p3;
    }
    // ---- rebuild P as two K=32 A-fragments ----
    Frag16 pa0, pa1;
    {
      const float* pr = myP + ln * PSTRIDE + 8 * g;
      float4 f0 = *(const float4*)(pr);
      float4 f1 = *(const float4*)(pr + 4);
      float4 f2 = *(const float4*)(pr + 16);
      float4 f3 = *(const float4*)(pr + 20);
      pa0.q[0] = make_uint4(pack2(f0.x,f0.y), pack2(f0.z,f0.w),
                            pack2(f1.x,f1.y), pack2(f1.z,f1.w));
      pa0.q[1] = make_uint4(pack2(f2.x,f2.y), pack2(f2.z,f2.w),
                            pack2(f3.x,f3.y), pack2(f3.z,f3.w));
      f0 = *(const float4*)(pr + 32);
      f1 = *(const float4*)(pr + 36);
      f2 = *(const float4*)(pr + 48);
      f3 = *(const float4*)(pr + 52);
      pa1.q[0] = make_uint4(pack2(f0.x,f0.y), pack2(f0.z,f0.w),
                            pack2(f1.x,f1.y), pack2(f1.z,f1.w));
      pa1.q[1] = make_uint4(pack2(f2.x,f2.y), pack2(f2.z,f2.w),
                            pack2(f3.x,f3.y), pack2(f3.z,f3.w));
    }
    // ---- O += P·V over 4 d-tiles, two K=32 halves ----
    #pragma unroll
    for (int t = 0; t < 4; ++t) {
      const unsigned short* vr = vbase + (size_t)(t * 16 + ln) * SEQ + l0 + 8 * g;
      Frag16 b;
      b.q[0] = *(const uint4*)(vr);
      b.q[1] = *(const uint4*)(vr + 16);
      o[t] = WMMA_BF16(pa0.v, b.v, o[t]);
      b.q[0] = *(const uint4*)(vr + 32);
      b.q[1] = *(const uint4*)(vr + 48);
      o[t] = WMMA_BF16(pa1.v, b.v, o[t]);
    }
  }
  // ---- normalize (one rcp per row) and store Y[n][k][h*64+d] ----
  float rinv[8];
  #pragma unroll
  for (int r = 0; r < 8; ++r) rinv[r] = 1.0f / lrow[r];
  unsigned short* yrow = Y + (size_t)(n * SEQ + q0) * (NHEADS * DHEAD) + h * DHEAD;
  #pragma unroll
  for (int t = 0; t < 4; ++t) {
    #pragma unroll
    for (int r = 0; r < 8; ++r) {
      yrow[(size_t)(r + 8 * g) * (NHEADS * DHEAD) + t * 16 + ln] = f2bf(o[t][r] * rinv[r]);
    }
  }
}

// ---------------------------------------------------------------------------
// Kernel 3: out = Y(bf16) @ Wo^T(bf16) + bo  -> f32 [NK][1024]
// ---------------------------------------------------------------------------
__global__ __launch_bounds__(128)
void out_proj_kernel(const unsigned short* __restrict__ Y,
                     const unsigned short* __restrict__ Wo,
                     const float* __restrict__ bo, float* __restrict__ out) {
  const int tid  = threadIdx.x;
  const int wave = tid >> 5, lane = tid & 31;
  const int ln = lane & 15, g = lane >> 4;
  const int q0 = blockIdx.x * 16;
  const int m0 = (blockIdx.y * 4 + wave) * 16;

  const unsigned short* yrow = Y  + (size_t)(q0 + ln) * DMODEL;
  const unsigned short* wrow = Wo + (size_t)(m0 + ln) * DMODEL;

  v8f acc = {0.f,0.f,0.f,0.f,0.f,0.f,0.f,0.f};
  #pragma unroll 4
  for (int k0 = 0; k0 < DMODEL; k0 += 32) {
    Frag16 a, b;
    a.q[0] = *(const uint4*)(yrow + k0 + 8 * g);
    a.q[1] = *(const uint4*)(yrow + k0 + 16 + 8 * g);
    b.q[0] = *(const uint4*)(wrow + k0 + 8 * g);
    b.q[1] = *(const uint4*)(wrow + k0 + 16 + 8 * g);
    acc = WMMA_BF16(a.v, b.v, acc);
  }
  const float bv = bo[m0 + ln];
  float* orow = out + (size_t)q0 * DMODEL + m0 + ln;
  #pragma unroll
  for (int r = 0; r < 8; ++r)
    orow[(size_t)(r + 8 * g) * DMODEL] = acc[r] + bv;
}

// ---------------------------------------------------------------------------
extern "C" void kernel_launch(void* const* d_in, const int* in_sizes, int n_in,
                              void* d_out, int out_size, void* d_ws, size_t ws_size,
                              hipStream_t stream) {
  const float* query = (const float*)d_in[0];
  const float* key_  = (const float*)d_in[1];
  const float* value = (const float*)d_in[2];
  const float* Wq    = (const float*)d_in[3];
  const float* bq    = (const float*)d_in[4];
  const float* Wk    = (const float*)d_in[5];
  const float* bk    = (const float*)d_in[6];
  const float* Wv    = (const float*)d_in[7];
  const float* bvv   = (const float*)d_in[8];
  const float* Wo    = (const float*)d_in[9];
  const float* bo    = (const float*)d_in[10];
  float* out = (float*)d_out;

  char* ws = (char*)d_ws;
  const size_t MB = 1024 * 1024;
  unsigned short* Qb  = (unsigned short*)(ws);             //  8 MB
  unsigned short* Kb  = (unsigned short*)(ws +  8 * MB);   //  8 MB
  unsigned short* Vt  = (unsigned short*)(ws + 16 * MB);   //  8 MB
  unsigned short* Yb  = (unsigned short*)(ws + 24 * MB);   //  8 MB
  unsigned short* Wqb = (unsigned short*)(ws + 32 * MB);   //  2 MB
  unsigned short* Wkb = (unsigned short*)(ws + 34 * MB);   //  2 MB
  unsigned short* Wvb = (unsigned short*)(ws + 36 * MB);   //  2 MB
  unsigned short* Wob = (unsigned short*)(ws + 38 * MB);   //  2 MB

  const int wN4 = (NHEADS * DHEAD * DMODEL) / 4;           // 262144
  cvt_bf16_kernel<<<wN4 / 256, 256, 0, stream>>>(Wq, Wqb, wN4);
  cvt_bf16_kernel<<<wN4 / 256, 256, 0, stream>>>(Wk, Wkb, wN4);
  cvt_bf16_kernel<<<wN4 / 256, 256, 0, stream>>>(Wv, Wvb, wN4);
  cvt_bf16_kernel<<<wN4 / 256, 256, 0, stream>>>(Wo, Wob, wN4);

  dim3 gProj(NKROWS / 16, NHEADS);
  const float qscale = 0.125f;   // 1/sqrt(DHEAD), folded into Q projection
  qkv_proj_kernel<<<gProj, 128, 0, stream>>>(query, Wqb, bq,  Qb, qscale, 0);
  qkv_proj_kernel<<<gProj, 128, 0, stream>>>(key_,  Wkb, bk,  Kb, 1.0f,   0);
  qkv_proj_kernel<<<gProj, 128, 0, stream>>>(value, Wvb, bvv, Vt, 1.0f,   1);
  attn_kernel<<<512, 256, 0, stream>>>(Qb, Kb, Vt, Yb);
  out_proj_kernel<<<dim3(NKROWS / 16, DMODEL / 64), 128, 0, stream>>>(Yb, Wob, bo, out);

  (void)in_sizes; (void)n_in; (void)out_size; (void)ws_size;
}
